// Perception_70489003262682
// MI455X (gfx1250) — matrix operand order repaired and checked by reading, not compiled, not run on hardware
//
#include <hip/hip_runtime.h>

// ---------------------------------------------------------------------------
// CDNA5 (gfx1250) implementation of the graph-transformer "Perception" module.
// All large GEMMs run on v_wmma_f32_16x16x32_bf16 (bf16 operands, f32 acc).
// Softmax / LayerNorm / reductions stay in f32.
// Data-movement plan (HBM-bound workload, 23.3 TB/s):
//   - adj (164 MB int32) read ONCE, coalesced, packed to a 41 MB u8 transposed
//     mask that stays L2-resident across the 3 GAT layers.
//   - H transposed per layer via LDS 32x32 tiles (coalesced both sides).
//   - All bf16 dot products use 16B vector loads.
// Input pointer order assumes setup_inputs() dict insertion order, nested
// dicts flattened depth-first in insertion order.
// ---------------------------------------------------------------------------

typedef unsigned short u16;
typedef __bf16 bf16_t;
typedef __attribute__((ext_vector_type(16))) __bf16 v16bf;
typedef __attribute__((ext_vector_type(8)))  __bf16 v8bf;
typedef __attribute__((ext_vector_type(8)))  float  v8f;
typedef __attribute__((ext_vector_type(8)))  unsigned short v8u;

#define BB 64
#define NN 800
#define DD 512
#define FFD 2048
#define NHEADS 8
#define DH 64
#define MAXT 510
#define NEGV (-1e9f)

__device__ __forceinline__ u16 f2bf(float f) {
  unsigned u = __float_as_uint(f);
  u += 0x7FFFu + ((u >> 16) & 1u);        // round-to-nearest-even
  return (u16)(u >> 16);
}
__device__ __forceinline__ float bf2f(u16 h) {
  return __uint_as_float(((unsigned)h) << 16);
}

__device__ __forceinline__ float blockSum(float v, float* red) {
  for (int o = 16; o; o >>= 1) v += __shfl_xor(v, o);
  if ((threadIdx.x & 31) == 0) red[threadIdx.x >> 5] = v;
  __syncthreads();
  float s = 0.f;
  int nw = (int)(blockDim.x >> 5);
  for (int i = 0; i < nw; ++i) s += red[i];
  __syncthreads();
  return s;
}
__device__ __forceinline__ float blockMax(float v, float* red) {
  for (int o = 16; o; o >>= 1) v = fmaxf(v, __shfl_xor(v, o));
  if ((threadIdx.x & 31) == 0) red[threadIdx.x >> 5] = v;
  __syncthreads();
  float s = -3.0e38f;
  int nw = (int)(blockDim.x >> 5);
  for (int i = 0; i < nw; ++i) s = fmaxf(s, red[i]);
  __syncthreads();
  return s;
}

// ---------------------------------------------------------------------------
// Generic bf16 WMMA GEMM:  C[M,N] = op(A[M,K] * Bcol + bias)
// A row-major bf16 (lda = K), B **column-major** bf16 (element (k,n) at n*K+k),
// C row-major (f32 or bf16). Batched via blockIdx.z with element strides.
// 4 waves per block; wave w computes rows [by*64 + w*16, +16) x cols [bx*64,+64).
// Requires: N % 64 == 0, K % 32 == 0 (true for all call sites here).
// EXEC stays full around every WMMA (loads clamp, only stores are guarded).
// ---------------------------------------------------------------------------
template <bool RELU, bool OUTBF>
__global__ __launch_bounds__(128) void k_gemm(
    const u16* __restrict__ Abuf, const u16* __restrict__ Bbuf,
    const float* __restrict__ bias, void* __restrict__ Cbuf,
    int M, int N, int K, long sA, long sB, long sC) {
  const bf16_t* A = reinterpret_cast<const bf16_t*>(Abuf) + (long)blockIdx.z * sA;
  const bf16_t* B = reinterpret_cast<const bf16_t*>(Bbuf) + (long)blockIdx.z * sB;
  const int wave = threadIdx.x >> 5;
  const int lane = threadIdx.x & 31;
  const int half = lane >> 4;
  const int l16  = lane & 15;
  const int rowBase = blockIdx.y * 64 + wave * 16;
  const int colBase = blockIdx.x * 64;

  int arow = rowBase + l16;
  if (arow >= M) arow = M - 1;               // clamp (stores are guarded)
  const bf16_t* Ar = A + (long)arow * K;

  v8f acc0, acc1, acc2, acc3;
  for (int i = 0; i < 8; ++i) { acc0[i] = 0.f; acc1[i] = 0.f; acc2[i] = 0.f; acc3[i] = 0.f; }

  const int n0 = colBase + l16;
  for (int k0 = 0; k0 < K; k0 += 32) {
    // A fragment: 16x32 bf16.  lane m = l16; VGPR pairs hold contiguous K runs.
    v16bf af;
    v8bf lo = *(const v8bf*)(Ar + k0 + 16 * half);
    v8bf hi = *(const v8bf*)(Ar + k0 + 16 * half + 8);
    for (int i = 0; i < 8; ++i) { af[i] = lo[i]; af[8 + i] = hi[i]; }
    // B fragments: 32x16 bf16, column-major -> one contiguous 32B load per lane.
    const bf16_t* Bk = B + k0 + 16 * half;
    v16bf b0 = *(const v16bf*)(Bk + (long)(n0)      * K);
    v16bf b1 = *(const v16bf*)(Bk + (long)(n0 + 16) * K);
    v16bf b2 = *(const v16bf*)(Bk + (long)(n0 + 32) * K);
    v16bf b3 = *(const v16bf*)(Bk + (long)(n0 + 48) * K);
    acc0 = __builtin_amdgcn_wmma_f32_16x16x32_bf16(false, af, false, b0, (short)0, acc0, false, false);
    acc1 = __builtin_amdgcn_wmma_f32_16x16x32_bf16(false, af, false, b1, (short)0, acc1, false, false);
    acc2 = __builtin_amdgcn_wmma_f32_16x16x32_bf16(false, af, false, b2, (short)0, acc2, false, false);
    acc3 = __builtin_amdgcn_wmma_f32_16x16x32_bf16(false, af, false, b3, (short)0, acc3, false, false);
  }

  const long cb = (long)blockIdx.z * sC;
  v8f accs[4] = {acc0, acc1, acc2, acc3};
  for (int t = 0; t < 4; ++t) {
    int n = colBase + t * 16 + l16;
    float bv = bias ? bias[n] : 0.f;
    for (int v = 0; v < 8; ++v) {
      int m = rowBase + v + 8 * half;
      if (m < M) {
        float val = accs[t][v] + bv;
        if (RELU) val = val > 0.f ? val : 0.f;
        if (OUTBF) ((u16*)Cbuf)[cb + (long)m * N + n] = f2bf(val);
        else       ((float*)Cbuf)[cb + (long)m * N + n] = val;
      }
    }
  }
}

// ---------------------------------------------------------------------------
// Elementwise / helper kernels
// ---------------------------------------------------------------------------

// Positional encoding table: pe[t, 2i] = sin(t*div_i), pe[t, 2i+1] = cos(t*div_i)
__global__ void k_pe(float* __restrict__ pe) {
  long idx = (long)blockIdx.x * blockDim.x + threadIdx.x;
  if (idx >= (long)MAXT * (DD / 2)) return;
  int t = (int)(idx >> 8);
  int i = (int)(idx & 255);
  float div = __expf(-(float)(2 * i) * (9.210340371976184f / (float)DD));
  float arg = (float)t * div;
  pe[(long)t * DD + 2 * i]     = sinf(arg);
  pe[(long)t * DD + 2 * i + 1] = cosf(arg);
}

// f32 row-major weight [K,N] -> bf16 column-major [N,K]
__global__ void k_wcol(const float* __restrict__ W, u16* __restrict__ out, int K, int N) {
  long idx = (long)blockIdx.x * blockDim.x + threadIdx.x;
  if (idx >= (long)K * N) return;
  int k = (int)(idx / N);
  int n = (int)(idx % N);
  out[(long)n * K + k] = f2bf(W[idx]);
}

__global__ void k_f2bf(const float* __restrict__ x, u16* __restrict__ y, long n) {
  long idx = (long)blockIdx.x * blockDim.x + threadIdx.x;
  if (idx < n) y[idx] = f2bf(x[idx]);
}

// cat([memory, broadcast goal]) -> bf16 [B*N, 2D]
__global__ void k_pack_cat(const float* __restrict__ mem, const float* __restrict__ goal,
                           u16* __restrict__ cat) {
  long idx = (long)blockIdx.x * blockDim.x + threadIdx.x;
  if (idx >= (long)BB * NN * (2 * DD)) return;
  int d = (int)(idx & (2 * DD - 1));
  long bn = idx >> 10;                 // /(2*DD)
  int b = (int)(bn / NN);
  float v;
  if (d < DD) v = mem[bn * DD + d];
  else        v = goal[(long)b * DD + (d - DD)];
  cat[idx] = f2bf(v);
}

// X_bf = bf16( Xf + PE[time[b,n]] )
__global__ void k_addpe(const float* __restrict__ Xf, const float* __restrict__ pe,
                        const int* __restrict__ gtime, u16* __restrict__ out) {
  long idx = (long)blockIdx.x * blockDim.x + threadIdx.x;
  if (idx >= (long)BB * NN * DD) return;
  long bn = idx >> 9;
  int d = (int)(idx & (DD - 1));
  int t = gtime[bn];
  out[idx] = f2bf(Xf[idx] + pe[(long)t * DD + d]);
}

// adj(int32)[b][s][t] -> transposed u8 mask mT[b][t][s], LDS 32x32 tiled so both
// the 164MB read and the 41MB write are fully coalesced. Runs ONCE.
__global__ __launch_bounds__(256) void k_maskT(const int* __restrict__ adj,
                                               unsigned char* __restrict__ mT) {
  __shared__ unsigned char tile[32][33];
  const int s0 = blockIdx.x * 32;
  const int t0 = blockIdx.y * 32;
  const long base = (long)blockIdx.z * NN * NN;
  for (int i = 0; i < 4; ++i) {
    int s = s0 + threadIdx.y + 8 * i;
    tile[threadIdx.y + 8 * i][threadIdx.x] =
        (unsigned char)(adj[base + (long)s * NN + t0 + threadIdx.x] > 0);
  }
  __syncthreads();
  for (int i = 0; i < 4; ++i) {
    int t = t0 + threadIdx.y + 8 * i;
    mT[base + (long)t * NN + s0 + threadIdx.x] = tile[threadIdx.x][threadIdx.y + 8 * i];
  }
}

// bf16 batched transpose via LDS 32x32 tiles: out[b][c][r] = in[b][r][c].
// R, C multiples of 32. grid (C/32, R/32, B), block (32,8).
__global__ __launch_bounds__(256) void k_transT(const u16* __restrict__ in,
                                                u16* __restrict__ out, int R, int C) {
  __shared__ u16 tile[32][33];
  const int r0 = blockIdx.y * 32;
  const int c0 = blockIdx.x * 32;
  const long base = (long)blockIdx.z * R * C;
  for (int i = 0; i < 4; ++i)
    tile[threadIdx.y + 8 * i][threadIdx.x] =
        in[base + (long)(r0 + threadIdx.y + 8 * i) * C + c0 + threadIdx.x];
  __syncthreads();
  for (int i = 0; i < 4; ++i)
    out[base + (long)(c0 + threadIdx.y + 8 * i) * R + r0 + threadIdx.x] =
        tile[threadIdx.x][threadIdx.y + 8 * i];
}

// ssrc[b,n] = <H[b,n,:], att_src>, sdst likewise. One wave per node; each lane
// owns 16 contiguous channels -> two 16B loads per lane.
__global__ void k_srcdst(const u16* __restrict__ H, const float* __restrict__ asr,
                         const float* __restrict__ adt, float* __restrict__ ssrc,
                         float* __restrict__ sdst) {
  long gid = (long)blockIdx.x * blockDim.x + threadIdx.x;
  long wid = gid >> 5;
  int lane = (int)(gid & 31);
  if (wid >= (long)BB * NN) return;
  const v8u* h8 = (const v8u*)(H + wid * DD);
  const int cbase = lane * 16;
  v8u a = h8[lane * 2];
  v8u c = h8[lane * 2 + 1];
  float s1 = 0.f, s2 = 0.f;
  for (int j = 0; j < 8; ++j) {
    float v = bf2f(a[j]);
    s1 += v * asr[cbase + j];
    s2 += v * adt[cbase + j];
  }
  for (int j = 0; j < 8; ++j) {
    float v = bf2f(c[j]);
    s1 += v * asr[cbase + 8 + j];
    s2 += v * adt[cbase + 8 + j];
  }
  for (int o = 16; o; o >>= 1) { s1 += __shfl_xor(s1, o); s2 += __shfl_xor(s2, o); }
  if (lane == 0) { ssrc[wid] = s1; sdst[wid] = s2; }
}

// alphaT[b][t][s] = softmax_s( mask(mT[b,t,s], lrelu(ssrc[s]+sdst[t])) ), bf16.
// mT is the pre-transposed u8 mask -> all reads coalesced (and L2-resident).
__global__ void k_alpha(const float* __restrict__ ssrc, const float* __restrict__ sdst,
                        const unsigned char* __restrict__ mT, u16* __restrict__ alphaT) {
  int t = blockIdx.x, b = blockIdx.y;
  __shared__ float ev[NN];
  __shared__ float red[8];
  const float st = sdst[(long)b * NN + t];
  const unsigned char* mrow = mT + ((long)b * NN + t) * NN;
  float lmax = -3.0e38f;
  for (int s = threadIdx.x; s < NN; s += blockDim.x) {
    float e;
    if (mrow[s]) {
      float z = ssrc[(long)b * NN + s] + st;
      e = (z > 0.f) ? z : 0.2f * z;
    } else e = NEGV;
    ev[s] = e;
    lmax = fmaxf(lmax, e);
  }
  float gmax = blockMax(lmax, red);
  float lsum = 0.f;
  for (int s = threadIdx.x; s < NN; s += blockDim.x) {
    float p = __expf(ev[s] - gmax);
    ev[s] = p;
    lsum += p;
  }
  float gsum = blockSum(lsum, red);
  float inv = 1.f / gsum;
  u16* arow = alphaT + ((long)b * NN + t) * NN;
  for (int s = threadIdx.x; s < NN; s += blockDim.x) arow[s] = f2bf(ev[s] * inv);
}

// Per-(b,h) masked softmax attention weights over N keys (single query token).
// K rows read as 16B vectors.
__global__ void k_scores(const float* __restrict__ Qf, const u16* __restrict__ Kp,
                         const unsigned char* __restrict__ mask, float* __restrict__ attn) {
  int h = blockIdx.x, b = blockIdx.y;
  __shared__ float sv[NN];
  __shared__ float qsh[DH];
  __shared__ float red[8];
  if (threadIdx.x < DH) qsh[threadIdx.x] = Qf[(long)b * DD + h * DH + threadIdx.x];
  __syncthreads();
  float lmax = -3.0e38f;
  for (int n = threadIdx.x; n < NN; n += blockDim.x) {
    const v8u* kr = (const v8u*)(Kp + ((long)b * NN + n) * DD + h * DH);
    float s = 0.f;
    for (int c = 0; c < 8; ++c) {
      v8u kk = kr[c];
      for (int j = 0; j < 8; ++j) s += qsh[c * 8 + j] * bf2f(kk[j]);
    }
    s *= 0.125f;                                   // 1/sqrt(64)
    if (!mask[(long)b * NN + n]) s = NEGV;
    sv[n] = s;
    lmax = fmaxf(lmax, s);
  }
  float gmax = blockMax(lmax, red);
  float lsum = 0.f;
  for (int n = threadIdx.x; n < NN; n += blockDim.x) {
    float p = __expf(sv[n] - gmax);
    sv[n] = p;
    lsum += p;
  }
  float gsum = blockSum(lsum, red);
  float inv = 1.f / gsum;
  for (int n = threadIdx.x; n < NN; n += blockDim.x)
    attn[((long)b * NHEADS + h) * NN + n] = sv[n] * inv;
}

// o[b, h*64+d] = sum_n attn[b,h,n] * V[b,n,h*64+d]
__global__ void k_attnout(const float* __restrict__ attn, const u16* __restrict__ Vp,
                          float* __restrict__ oF) {
  int h = blockIdx.x, b = blockIdx.y;
  int d = threadIdx.x;                               // 64 threads
  const float* a = attn + ((long)b * NHEADS + h) * NN;
  const u16* vb = Vp + (long)b * NN * DD + h * DH + d;
  float acc = 0.f;
  for (int n = 0; n < NN; ++n) acc += a[n] * bf2f(vb[(long)n * DD]);
  oF[(long)b * DD + h * DH + d] = acc;
}

// out = LayerNorm(x + r) * g + b ; optional bf16 copy. block=256, D=512, row=blockIdx.x
__global__ void k_ln(const float* __restrict__ x, const float* __restrict__ r,
                     const float* __restrict__ g, const float* __restrict__ bp,
                     float* __restrict__ outF, u16* __restrict__ outBF) {
  int row = blockIdx.x;
  __shared__ float red[8];
  int c0 = threadIdx.x, c1 = threadIdx.x + 256;
  const float* xr = x + (long)row * DD;
  const float* rr = r + (long)row * DD;
  float v0 = xr[c0] + rr[c0];
  float v1 = xr[c1] + rr[c1];
  float mean = blockSum(v0 + v1, red) * (1.f / DD);
  float d0 = v0 - mean, d1 = v1 - mean;
  float var = blockSum(d0 * d0 + d1 * d1, red) * (1.f / DD);
  float inv = rsqrtf(var + 1e-5f);
  float o0 = d0 * inv * g[c0] + bp[c0];
  float o1 = d1 * inv * g[c1] + bp[c1];
  outF[(long)row * DD + c0] = o0;
  outF[(long)row * DD + c1] = o1;
  if (outBF) {
    outBF[(long)row * DD + c0] = f2bf(o0);
    outBF[(long)row * DD + c1] = f2bf(o1);
  }
}

// ---------------------------------------------------------------------------
// Host-side helpers
// ---------------------------------------------------------------------------
static void launch_gemm(hipStream_t st, const void* A, const void* B, const float* bias,
                        void* C, int M, int N, int K, long sA, long sB, long sC, int batch,
                        bool relu, bool outbf) {
  dim3 g(N / 64, (M + 63) / 64, batch);
  dim3 b(128);
  const u16* Ap = (const u16*)A;
  const u16* Bp = (const u16*)B;
  if (relu && outbf)
    k_gemm<true, true><<<g, b, 0, st>>>(Ap, Bp, bias, C, M, N, K, sA, sB, sC);
  else if (relu)
    k_gemm<true, false><<<g, b, 0, st>>>(Ap, Bp, bias, C, M, N, K, sA, sB, sC);
  else if (outbf)
    k_gemm<false, true><<<g, b, 0, st>>>(Ap, Bp, bias, C, M, N, K, sA, sB, sC);
  else
    k_gemm<false, false><<<g, b, 0, st>>>(Ap, Bp, bias, C, M, N, K, sA, sB, sC);
}

static inline int cdiv(long n, int d) { return (int)((n + d - 1) / d); }

extern "C" void kernel_launch(void* const* d_in, const int* in_sizes, int n_in,
                              void* d_out, int out_size, void* d_ws, size_t ws_size,
                              hipStream_t stream) {
  // ---- input pointers (setup_inputs() dict insertion order, nested depth-first)
  const float* mem   = (const float*)d_in[0];   // [B,N,D]
  const float* goalE = (const float*)d_in[1];   // [B,D]
  const float* currE = (const float*)d_in[2];   // [B,D]
  const int*   adj   = (const int*)d_in[3];     // [B,N,N]
  const int*   gtime = (const int*)d_in[4];     // [B,N]
  const unsigned char* gmask = (const unsigned char*)d_in[5];  // [B,N] bool
  const float* feW1 = (const float*)d_in[6];    // [2D,D]
  const float* feB1 = (const float*)d_in[7];
  const float* feW2 = (const float*)d_in[8];    // [D,D]
  const float* feB2 = (const float*)d_in[9];
  const float *gatW[3], *gatAS[3], *gatAD[3], *gatB[3];
  for (int l = 0; l < 3; ++l) {
    gatW[l]  = (const float*)d_in[10 + 4 * l];
    gatAS[l] = (const float*)d_in[11 + 4 * l];
    gatAD[l] = (const float*)d_in[12 + 4 * l];
    gatB[l]  = (const float*)d_in[13 + 4 * l];
  }
  const int pbase[2] = {22, 38};  // goal_dec, curr_dec param bases

  // ---- workspace layout (all 256B aligned). Total ~405 MB.
  char* base = (char*)d_ws;
  size_t off = 0;
  auto alloc = [&](size_t bytes) -> void* {
    void* p = base + off;
    off = (off + bytes + 255) & ~(size_t)255;
    return p;
  };
  float* peF  = (float*)alloc((size_t)MAXT * DD * 4);
  u16* wfe1   = (u16*)alloc((size_t)2 * DD * DD * 2);
  u16* wfe2   = (u16*)alloc((size_t)DD * DD * 2);
  u16* wgat[3];
  for (int l = 0; l < 3; ++l) wgat[l] = (u16*)alloc((size_t)DD * DD * 2);
  u16 *dwq[2], *dwk[2], *dwv[2], *dwo[2], *dwl1[2], *dwl2[2];
  for (int dx = 0; dx < 2; ++dx) {
    dwq[dx]  = (u16*)alloc((size_t)DD * DD * 2);
    dwk[dx]  = (u16*)alloc((size_t)DD * DD * 2);
    dwv[dx]  = (u16*)alloc((size_t)DD * DD * 2);
    dwo[dx]  = (u16*)alloc((size_t)DD * DD * 2);
    dwl1[dx] = (u16*)alloc((size_t)DD * FFD * 2);
    dwl2[dx] = (u16*)alloc((size_t)FFD * DD * 2);
  }
  float* ssrc = (float*)alloc((size_t)BB * NN * 4);
  float* sdst = (float*)alloc((size_t)BB * NN * 4);
  u16*   qembBF = (u16*)alloc((size_t)BB * DD * 2);
  float* Qf     = (float*)alloc((size_t)BB * DD * 4);
  float* attnP  = (float*)alloc((size_t)BB * NHEADS * NN * 4);
  float* oF     = (float*)alloc((size_t)BB * DD * 4);
  u16*   oBF    = (u16*)alloc((size_t)BB * DD * 2);
  float* oproj  = (float*)alloc((size_t)BB * DD * 4);
  float* xF     = (float*)alloc((size_t)BB * DD * 4);
  u16*   xBF    = (u16*)alloc((size_t)BB * DD * 2);
  u16*   ff1    = (u16*)alloc((size_t)BB * FFD * 2);
  float* ff2    = (float*)alloc((size_t)BB * DD * 4);
  // big buffers
  u16* bufCat = (u16*)alloc((size_t)BB * NN * 2 * DD * 2);  // 105 MB (also reused as f32 X)
  u16* buf1   = (u16*)alloc((size_t)BB * NN * DD * 2);      // 52 MB (X1 / Ht / V)
  u16* buf2   = (u16*)alloc((size_t)BB * NN * DD * 2);      // 52 MB (X / ctx)
  u16* buf3   = (u16*)alloc((size_t)BB * NN * DD * 2);      // 52 MB (H / K)
  u16* alphaT = (u16*)alloc((size_t)BB * NN * NN * 2);      // 82 MB
  unsigned char* maskT = (unsigned char*)alloc((size_t)BB * NN * NN);  // 41 MB
  (void)ws_size; (void)in_sizes; (void)n_in; (void)out_size;

  const long BND = (long)BB * NN * DD;

  // ---- constants & weight conversion (f32 -> bf16 column-major)
  k_pe<<<cdiv((long)MAXT * (DD / 2), 256), 256, 0, stream>>>(peF);
  k_wcol<<<cdiv((long)2 * DD * DD, 256), 256, 0, stream>>>(feW1, wfe1, 2 * DD, DD);
  k_wcol<<<cdiv((long)DD * DD, 256), 256, 0, stream>>>(feW2, wfe2, DD, DD);
  for (int l = 0; l < 3; ++l)
    k_wcol<<<cdiv((long)DD * DD, 256), 256, 0, stream>>>(gatW[l], wgat[l], DD, DD);
  for (int dx = 0; dx < 2; ++dx) {
    int pb = pbase[dx];
    k_wcol<<<cdiv((long)DD * DD, 256), 256, 0, stream>>>((const float*)d_in[pb + 0], dwq[dx], DD, DD);
    k_wcol<<<cdiv((long)DD * DD, 256), 256, 0, stream>>>((const float*)d_in[pb + 2], dwk[dx], DD, DD);
    k_wcol<<<cdiv((long)DD * DD, 256), 256, 0, stream>>>((const float*)d_in[pb + 4], dwv[dx], DD, DD);
    k_wcol<<<cdiv((long)DD * DD, 256), 256, 0, stream>>>((const float*)d_in[pb + 6], dwo[dx], DD, DD);
    k_wcol<<<cdiv((long)DD * FFD, 256), 256, 0, stream>>>((const float*)d_in[pb + 8], dwl1[dx], DD, FFD);
    k_wcol<<<cdiv((long)FFD * DD, 256), 256, 0, stream>>>((const float*)d_in[pb + 10], dwl2[dx], FFD, DD);
  }
  // adjacency -> transposed u8 mask (single coalesced pass over the 164MB adj)
  k_maskT<<<dim3(NN / 32, NN / 32, BB), dim3(32, 8), 0, stream>>>(adj, maskT);

  // ---- feature embedding: X = relu(cat @ W1 + b1) @ W2 + b2 + PE[time]
  k_pack_cat<<<cdiv((long)BB * NN * 2 * DD, 256), 256, 0, stream>>>(mem, goalE, bufCat);
  launch_gemm(stream, bufCat, wfe1, feB1, buf1, BB * NN, DD, 2 * DD, 0, 0, 0, 1, true, true);
  launch_gemm(stream, buf1, wfe2, feB2, (float*)bufCat, BB * NN, DD, DD, 0, 0, 0, 1, false, false);
  k_addpe<<<cdiv(BND, 256), 256, 0, stream>>>((const float*)bufCat, peF, gtime, buf2);

  // ---- 3x GAT layers
  for (int l = 0; l < 3; ++l) {
    launch_gemm(stream, buf2, wgat[l], nullptr, buf3, BB * NN, DD, DD, 0, 0, 0, 1, false, true);
    k_srcdst<<<cdiv((long)BB * NN * 32, 256), 256, 0, stream>>>(buf3, gatAS[l], gatAD[l], ssrc, sdst);
    k_alpha<<<dim3(NN, BB), 256, 0, stream>>>(ssrc, sdst, maskT, alphaT);
    k_transT<<<dim3(DD / 32, NN / 32, BB), dim3(32, 8), 0, stream>>>(buf3, buf1, NN, DD);
    // X' = alphaT @ H + bias  (batched over B)
    launch_gemm(stream, alphaT, buf1, gatB[l], buf2, NN, DD, NN,
                (long)NN * NN, (long)DD * NN, (long)NN * DD, BB, false, true);
  }
  // ctx (bf16) now in buf2

  // ---- decoder blocks (single query token each)
  auto run_dec = [&](int dx, const float* embF, float* outPtr) {
    int pb = pbase[dx];
    const float* bq  = (const float*)d_in[pb + 1];
    const float* bk  = (const float*)d_in[pb + 3];
    const float* bv  = (const float*)d_in[pb + 5];
    const float* bo  = (const float*)d_in[pb + 7];
    const float* l1b = (const float*)d_in[pb + 9];
    const float* l2b = (const float*)d_in[pb + 11];
    const float* g1  = (const float*)d_in[pb + 12];
    const float* b1  = (const float*)d_in[pb + 13];
    const float* g2  = (const float*)d_in[pb + 14];
    const float* b2  = (const float*)d_in[pb + 15];
    k_f2bf<<<cdiv((long)BB * DD, 256), 256, 0, stream>>>(embF, qembBF, (long)BB * DD);
    launch_gemm(stream, qembBF, dwq[dx], bq, Qf, BB, DD, DD, 0, 0, 0, 1, false, false);
    launch_gemm(stream, buf2, dwk[dx], bk, buf3, BB * NN, DD, DD, 0, 0, 0, 1, false, true);
    launch_gemm(stream, buf2, dwv[dx], bv, buf1, BB * NN, DD, DD, 0, 0, 0, 1, false, true);
    k_scores<<<dim3(NHEADS, BB), 256, 0, stream>>>(Qf, buf3, gmask, attnP);
    k_attnout<<<dim3(NHEADS, BB), DH, 0, stream>>>(attnP, buf1, oF);
    k_f2bf<<<cdiv((long)BB * DD, 256), 256, 0, stream>>>(oF, oBF, (long)BB * DD);
    launch_gemm(stream, oBF, dwo[dx], bo, oproj, BB, DD, DD, 0, 0, 0, 1, false, false);
    k_ln<<<BB, 256, 0, stream>>>(embF, oproj, g1, b1, xF, xBF);
    launch_gemm(stream, xBF, dwl1[dx], l1b, ff1, BB, FFD, DD, 0, 0, 0, 1, true, true);
    launch_gemm(stream, ff1, dwl2[dx], l2b, ff2, BB, DD, FFD, 0, 0, 0, 1, false, false);
    k_ln<<<BB, 256, 0, stream>>>(xF, ff2, g2, b2, outPtr, nullptr);
  };

  // output stack: out[0] = curr_ctx, out[1] = goal_ctx
  run_dec(1, currE, (float*)d_out);
  run_dec(0, goalE, (float*)d_out + (long)BB * DD);
}
